// BiLSTM_CRF_31808527794635
// MI455X (gfx1250) — matrix operand order repaired and checked by reading, not compiled, not run on hardware
//
#include <hip/hip_runtime.h>
#include <cstdint>

// ---------------------------------------------------------------------------
// BiLSTM-CRF forward loss for MI455X (gfx1250, wave32, WMMA).
// B=32, T=256, V=50000, E=256, H=512, K=24, L=2.
//
// Strategy:
//  * All dense math in bf16 WMMA (v_wmma_f32_16x16x32_bf16), f32 accumulate.
//  * Weights converted once per launch to bf16 (19 MB, L2-resident).
//  * Input projections xg = x @ W_ih^T + b : big parallel WMMA GEMMs,
//    2 M-tiles per wave (B-fragment register reuse), branchless ragged-N
//    handling via a zero-row pointer select (no LDS temporaries in the loop).
//  * Recurrence: persistent cooperative kernel, 16 blocks x 256 threads.
//    Each block stages a 128 KB bf16 W_hh slice in LDS (CDNA5: 320 KB/WGP),
//    computes gates with WMMA, double-buffers h (bf16, 32 KB) through L2,
//    grid barrier via agent-scope atomics (one barrier/step).
//  * CRF: one block, wave-per-batch, shuffle logsumexp.
// ---------------------------------------------------------------------------

#define Bq 32
#define Tq 256
#define Eq 256
#define Hq 512
#define Kq 24
#define G4H 2048           // 4*H
#define MROWS (Tq * Bq)    // 8192, time-major rows (t*B + b)

typedef __attribute__((ext_vector_type(16))) __bf16 v16bf;
typedef __attribute__((ext_vector_type(8)))  float  v8f;

struct Frag32 { uint4 lo, hi; };   // 32 bytes == 16 bf16 elements

static __device__ __forceinline__ v16bf frag_cast(const Frag32& f) {
  return __builtin_bit_cast(v16bf, f);
}

static __device__ __forceinline__ unsigned short f32_to_bf16(float f) {
  unsigned int u = __builtin_bit_cast(unsigned int, f);
  u += 0x7FFFu + ((u >> 16) & 1u);       // round-to-nearest-even
  return (unsigned short)(u >> 16);
}

static __device__ __forceinline__ float sigmoidf_(float x) {
  return 1.0f / (1.0f + __expf(-x));
}

// A fragment (16-bit A 16x32): lane L<16 holds row M=L,
// K {kb..kb+15}; lane 16+L holds row L, K {kb+16..kb+31} (matches B).
// 16 contiguous bf16 in row-major [M, lda] storage.
static __device__ __forceinline__ v16bf load_a_frag(const unsigned short* A, int lda,
                                                    int mbase, int kbase, int lane) {
  int row = mbase + (lane & 15);
  int hs  = (lane >> 4) * 16;
  const unsigned short* p = A + (size_t)row * lda + kbase + hs;
  Frag32 f;
  f.lo = *(const uint4*)(p);
  f.hi = *(const uint4*)(p + 8);
  return frag_cast(f);
}

// B fragment (16-bit B 32x16): lane L holds column n = L%16 of W^T,
// K range (L<16 ? kb..kb+15 : kb+16..kb+31) -> 16 contiguous bf16 from a
// row pointer (already bounds-selected by caller; branchless).
static __device__ __forceinline__ v16bf load_b_row(const unsigned short* brow,
                                                   int kbase, int lane) {
  int ko = kbase + (lane >> 4) * 16;
  Frag32 f;
  f.lo = *(const uint4*)(brow + ko);
  f.hi = *(const uint4*)(brow + ko + 8);
  return frag_cast(f);
}

// ---------------------------------------------------------------------------
// generic elementwise fp32 -> bf16 convert
// ---------------------------------------------------------------------------
__global__ void f2bf_kernel(const float* __restrict__ src,
                            unsigned short* __restrict__ dst, int n) {
  for (int i = blockIdx.x * blockDim.x + threadIdx.x; i < n;
       i += gridDim.x * blockDim.x)
    dst[i] = f32_to_bf16(src[i]);
}

__global__ void init_ints_kernel(int* p, int n) {
  int i = blockIdx.x * blockDim.x + threadIdx.x;
  if (i < n) p[i] = 0;
}

// ---------------------------------------------------------------------------
// embedding gather (time-major rows t*B + b), ld = 1024; also mask.
// ids layout [B, T]. embedding row 0 is zero (matches reference).
// ---------------------------------------------------------------------------
__global__ void embed_kernel(const int* __restrict__ ids,
                             const float* __restrict__ emb,
                             unsigned short* __restrict__ xin,   // [8192, 1024]
                             float* __restrict__ maskf) {        // [8192]
  int tb = blockIdx.x;            // t*B + b
  int b  = tb & (Bq - 1);
  int t  = tb >> 5;
  int id = ids[b * Tq + t];
  int e  = threadIdx.x;           // 256 threads = E
  float v = emb[(size_t)id * Eq + e];
  xin[(size_t)tb * 1024 + e] = f32_to_bf16(v);
  if (e == 0) maskf[tb] = (id != 0) ? 1.0f : 0.0f;
}

// ---------------------------------------------------------------------------
// bf16 WMMA GEMM: out[M,N] = A[M,K] * W[N,K]^T + bias.
// 8 waves/block, each wave computes a 32x16 output tile (2 M-tiles sharing
// one B fragment). Ragged N handled with a zero-row pointer select.
// ---------------------------------------------------------------------------
__global__ void gemm_bf16_kernel(const unsigned short* __restrict__ A, int lda,
                                 const unsigned short* __restrict__ W, int ldw,
                                 const float* __restrict__ bias,
                                 float* __restrict__ out, int ldo,
                                 int M, int N, int K,
                                 const unsigned short* __restrict__ zrow) {
  int lane  = threadIdx.x & 31;
  int wave  = threadIdx.x >> 5;
  int mbase = blockIdx.x * 32;
  int nbase = (blockIdx.y * 8 + wave) * 16;
  if (nbase >= N) return;                      // wave-uniform
  int n = nbase + (lane & 15);
  const unsigned short* brow = (n < N) ? (W + (size_t)n * ldw) : zrow;
  v8f acc0 = {}, acc1 = {};
  for (int k = 0; k < K; k += 32) {
    if (k + 32 < K)
      __builtin_prefetch(A + (size_t)(mbase + (lane & 15)) * lda + k + 32, 0, 1);
    v16bf b  = load_b_row(brow, k, lane);
    v16bf a0 = load_a_frag(A, lda, mbase,      k, lane);
    v16bf a1 = load_a_frag(A, lda, mbase + 16, k, lane);
    acc0 = __builtin_amdgcn_wmma_f32_16x16x32_bf16(false, a0, false, b,
                                                   (short)0, acc0, false, false);
    acc1 = __builtin_amdgcn_wmma_f32_16x16x32_bf16(false, a1, false, b,
                                                   (short)0, acc1, false, false);
  }
  // C/D layout: lane<16 -> (M = r,  N = lane); lane>=16 -> (M = 8+r, N = lane-16)
  if (n >= N) return;
  float bv = bias ? bias[n] : 0.0f;
  int mrow = mbase + (lane >> 4) * 8;
  for (int r = 0; r < 8; ++r)
    out[(size_t)(mrow + r) * ldo + n] = acc0[r] + bv;
  for (int r = 0; r < 8; ++r)
    out[(size_t)(mrow + 16 + r) * ldo + n] = acc1[r] + bv;
}

// ---------------------------------------------------------------------------
// grid-wide barrier (agent scope)
// ---------------------------------------------------------------------------
static __device__ __forceinline__ void grid_sync(int* ctr, int* gen, int nblk) {
  __syncthreads();
  if (threadIdx.x == 0) {
    __threadfence();
    int g = __hip_atomic_load(gen, __ATOMIC_ACQUIRE, __HIP_MEMORY_SCOPE_AGENT);
    int a = __hip_atomic_fetch_add(ctr, 1, __ATOMIC_ACQ_REL, __HIP_MEMORY_SCOPE_AGENT);
    if (a == nblk - 1) {
      __hip_atomic_store(ctr, 0, __ATOMIC_RELAXED, __HIP_MEMORY_SCOPE_AGENT);
      __hip_atomic_fetch_add(gen, 1, __ATOMIC_ACQ_REL, __HIP_MEMORY_SCOPE_AGENT);
    } else {
      while (__hip_atomic_load(gen, __ATOMIC_ACQUIRE, __HIP_MEMORY_SCOPE_AGENT) == g)
        __builtin_amdgcn_s_sleep(2);
    }
  }
  __syncthreads();
}

// ---------------------------------------------------------------------------
// persistent LSTM scan. 16 blocks x 256 threads (8 waves).
// Block b owns h columns [32b, 32b+32) across all 4 gates (128 gate cols).
// Wave w -> gate g=w/2, 16-col subtile (w&1). W_hh slice lives in LDS.
// Dynamic LDS: 128K (W) + 16K (gates) + 4K (c) + 4K (h) = 152 KB.
// ---------------------------------------------------------------------------
#define SCAN_BLOCKS 16
#define SCAN_LDS_BYTES (152 * 1024)

__global__ void lstm_scan_kernel(const float* __restrict__ xg,          // [8192, 2048]
                                 const unsigned short* __restrict__ Whh, // [2048, 512] bf16
                                 unsigned short* __restrict__ h0buf,     // [32, 512] bf16
                                 unsigned short* __restrict__ h1buf,
                                 const float* __restrict__ maskf,        // [8192]
                                 float* __restrict__ y, int ycol0,       // [8192, 1024]
                                 int reverse, int* ctr, int* gen) {
  extern __shared__ char smem[];
  unsigned short* ldsW = (unsigned short*)smem;              // 8 tiles * 16 rows * 512 K
  float* ldsG = (float*)(smem + 128 * 1024);                 // [4][32][32]
  float* ldsC = (float*)(smem + 144 * 1024);                 // [32][32]
  float* ldsH = (float*)(smem + 148 * 1024);                 // [32][32]

  int tid  = threadIdx.x;
  int lane = tid & 31;
  int wave = tid >> 5;
  int blk  = blockIdx.x;

  // ---- preload this block's W_hh slice: 128 rows x 512 K (bf16) into LDS
  for (int idx = tid; idx < 8 * 16 * 64; idx += 256) {       // 64 uint4 per row
    int rowid = idx >> 6;
    int chunk = idx & 63;
    int tile  = rowid >> 4;
    int r     = rowid & 15;
    int gg    = tile >> 1;
    int n     = gg * Hq + blk * 32 + (tile & 1) * 16 + r;
    ((uint4*)ldsW)[rowid * 64 + chunk] =
        *(const uint4*)(Whh + (size_t)n * Hq + chunk * 8);
  }
  // ---- zero c/h state and our slice of both h global buffers
  for (int idx = tid; idx < 1024; idx += 256) { ldsC[idx] = 0.f; ldsH[idx] = 0.f; }
  for (int idx = tid; idx < 1024; idx += 256) {
    int m = idx >> 5, j = idx & 31;
    h0buf[m * Hq + blk * 32 + j] = 0;
    h1buf[m * Hq + blk * 32 + j] = 0;
  }
  grid_sync(ctr, gen, SCAN_BLOCKS);

  int g    = wave >> 1;
  int ncol = g * Hq + blk * 32 + (wave & 1) * 16;            // gate-space column base

  for (int step = 0; step < Tq; ++step) {
    int t = reverse ? (Tq - 1 - step) : step;
    const unsigned short* hread = (step & 1) ? h1buf : h0buf;
    unsigned short*       hwrit = (step & 1) ? h0buf : h1buf;

    // ---- gates(:,ncol..ncol+15) = h(prev) @ Whh^T ; two 16x16 M-tiles
    v8f acc0 = {}, acc1 = {};
    for (int kk = 0; kk < Hq; kk += 32) {
      int lr = wave * 16 + (lane & 15);
      int ko = kk + (lane >> 4) * 16;
      Frag32 fb;
      const uint4* bp = (const uint4*)(ldsW + (size_t)lr * Hq + ko);
      fb.lo = bp[0]; fb.hi = bp[1];
      v16bf b  = frag_cast(fb);
      v16bf a0 = load_a_frag(hread, Hq, 0,  kk, lane);
      v16bf a1 = load_a_frag(hread, Hq, 16, kk, lane);
      acc0 = __builtin_amdgcn_wmma_f32_16x16x32_bf16(false, a0, false, b,
                                                     (short)0, acc0, false, false);
      acc1 = __builtin_amdgcn_wmma_f32_16x16x32_bf16(false, a1, false, b,
                                                     (short)0, acc1, false, false);
    }

    // ---- add xg and exchange gates through LDS: ldsG[g][m][jl]
    {
      int n  = ncol + (lane & 15);
      int jl = (wave & 1) * 16 + (lane & 15);
      int m0 = (lane >> 4) * 8;
      const float* gx = xg + (size_t)t * Bq * G4H;
      for (int r = 0; r < 8; ++r) {
        ldsG[g * 1024 + (m0 + r) * 32 + jl]      = acc0[r] + gx[(size_t)(m0 + r) * G4H + n];
        ldsG[g * 1024 + (16 + m0 + r) * 32 + jl] = acc1[r] + gx[(size_t)(16 + m0 + r) * G4H + n];
      }
    }
    __syncthreads();

    // ---- elementwise LSTM update for this block's 32x32 (batch x hcol) slice
    for (int idx = tid; idx < 1024; idx += 256) {
      int m = idx >> 5, j = idx & 31;
      float iv = ldsG[0 * 1024 + idx];
      float fv = ldsG[1 * 1024 + idx];
      float gv = ldsG[2 * 1024 + idx];
      float ov = ldsG[3 * 1024 + idx];
      float c_old = ldsC[idx];
      float h_old = ldsH[idx];
      float cn = sigmoidf_(fv) * c_old + sigmoidf_(iv) * tanhf(gv);
      float hn = sigmoidf_(ov) * tanhf(cn);
      float mv = maskf[t * Bq + m];
      float h2 = mv * hn + (1.0f - mv) * h_old;
      float c2 = mv * cn + (1.0f - mv) * c_old;
      ldsC[idx] = c2;
      ldsH[idx] = h2;
      y[(size_t)(t * Bq + m) * 1024 + ycol0 + blk * 32 + j] = mv * hn;
      hwrit[m * Hq + blk * 32 + j] = f32_to_bf16(h2);
    }
    grid_sync(ctr, gen, SCAN_BLOCKS);          // h(write) visible everywhere
  }
}

// ---------------------------------------------------------------------------
// CRF: score + forward-algorithm logZ + loss. One block, wave b = batch b.
// ---------------------------------------------------------------------------
__global__ void crf_kernel(const float* __restrict__ em,      // [8192, 24]
                           const float* __restrict__ maskf,   // [8192]
                           const int* __restrict__ tags,      // [B, T]
                           const int* __restrict__ lengths,   // [B]
                           const float* __restrict__ startv,
                           const float* __restrict__ endv,
                           const float* __restrict__ transv,  // [24,24]
                           float* __restrict__ out) {
  __shared__ float ltrans[Kq * Kq];
  __shared__ float lstart[Kq], lend[Kq];
  __shared__ float perb[Bq];
  int tid  = threadIdx.x;
  int lane = tid & 31;
  int b    = tid >> 5;                 // 32 waves, one per batch element
  if (tid < Kq * Kq) ltrans[tid] = transv[tid];
  if (tid < Kq) { lstart[tid] = startv[tid]; lend[tid] = endv[tid]; }
  __syncthreads();

  // ---- forward algorithm (lane j holds alpha_j)
  float alpha = (lane < Kq) ? (lstart[lane] + em[(size_t)(0 * Bq + b) * Kq + lane])
                            : -1e30f;
  for (int t = 1; t < Tq; ++t) {
    float best = -1e30f;
    for (int i = 0; i < Kq; ++i) {
      float ai = __shfl(alpha, i, 32);
      float s  = ai + ((lane < Kq) ? ltrans[i * Kq + lane] : 0.0f);
      best = fmaxf(best, s);
    }
    float sum = 0.0f;
    for (int i = 0; i < Kq; ++i) {
      float ai = __shfl(alpha, i, 32);
      float s  = ai + ((lane < Kq) ? ltrans[i * Kq + lane] : 0.0f);
      sum += __expf(s - best);
    }
    float nxt = best + __logf(sum) + em[(size_t)(t * Bq + b) * Kq + (lane < Kq ? lane : 0)];
    float mv  = maskf[t * Bq + b];
    if (lane < Kq) alpha = (mv > 0.0f) ? nxt : alpha;
  }
  // logZ = logsumexp(alpha + end)
  float v = (lane < Kq) ? (alpha + lend[lane]) : -1e30f;
  float mx = v;
  for (int off = 16; off; off >>= 1) mx = fmaxf(mx, __shfl_xor(mx, off, 32));
  float e = (lane < Kq) ? __expf(v - mx) : 0.0f;
  for (int off = 16; off; off >>= 1) e += __shfl_xor(e, off, 32);
  float logZ = mx + __logf(e);

  // ---- gold path score (serial, lane 0)
  if (lane == 0) {
    int prev = tags[b * Tq + 0];
    float s = lstart[prev] + em[(size_t)(0 * Bq + b) * Kq + prev];
    for (int t = 1; t < Tq; ++t) {
      int tg   = tags[b * Tq + t];
      float mv = maskf[t * Bq + b];
      s += mv * (ltrans[prev * Kq + tg] + em[(size_t)(t * Bq + b) * Kq + tg]);
      prev = tg;
    }
    int last = tags[b * Tq + (lengths[b] - 1)];
    s += lend[last];
    perb[b] = s - logZ;
  }
  __syncthreads();
  if (tid == 0) {
    float s = 0.0f;
    for (int i = 0; i < Bq; ++i) s += perb[i];
    out[0] = -s / (float)Bq;
  }
}

// ---------------------------------------------------------------------------
// host orchestration
// ---------------------------------------------------------------------------
extern "C" void kernel_launch(void* const* d_in, const int* in_sizes, int n_in,
                              void* d_out, int out_size, void* d_ws, size_t ws_size,
                              hipStream_t stream) {
  (void)in_sizes; (void)n_in; (void)out_size; (void)ws_size;

  const int*   ids      = (const int*)d_in[0];
  const int*   lengths  = (const int*)d_in[1];
  const int*   tags     = (const int*)d_in[2];
  const float* emb      = (const float*)d_in[3];
  const float* Wih_f[2][2] = {{(const float*)d_in[4],  (const float*)d_in[7]},
                              {(const float*)d_in[10], (const float*)d_in[13]}};
  const float* Whh_f[2][2] = {{(const float*)d_in[5],  (const float*)d_in[8]},
                              {(const float*)d_in[11], (const float*)d_in[14]}};
  const float* bias_f[2][2] = {{(const float*)d_in[6],  (const float*)d_in[9]},
                               {(const float*)d_in[12], (const float*)d_in[15]}};
  const float* fcW   = (const float*)d_in[16];
  const float* fcb   = (const float*)d_in[17];
  const float* crf_s = (const float*)d_in[18];
  const float* crf_e = (const float*)d_in[19];
  const float* crf_t = (const float*)d_in[20];

  char* ws = (char*)d_ws;
  size_t off = 0;
  auto alloc = [&](size_t bytes) -> char* {
    char* p = ws + off;
    off = (off + bytes + 255) & ~(size_t)255;
    return p;
  };

  int*            barrier = (int*)           alloc(256);
  unsigned short* zrow    = (unsigned short*)alloc(1024 * 2);   // zero row, K<=1024
  unsigned short* xin     = (unsigned short*)alloc((size_t)MROWS * 1024 * 2);  // 16 MB
  float*          xg      = (float*)         alloc((size_t)MROWS * G4H * 4);   // 64 MB
  float*          ybuf    = (float*)         alloc((size_t)MROWS * 1024 * 4);  // 32 MB
  float*          embuf   = (float*)         alloc((size_t)MROWS * Kq * 4);
  float*          maskf   = (float*)         alloc((size_t)MROWS * 4);
  unsigned short* h0      = (unsigned short*)alloc((size_t)Bq * Hq * 2);
  unsigned short* h1      = (unsigned short*)alloc((size_t)Bq * Hq * 2);
  unsigned short* wih_b[2][2];
  unsigned short* whh_b[2][2];
  const int din[2] = {Eq, 2 * Hq};
  for (int l = 0; l < 2; ++l)
    for (int d = 0; d < 2; ++d) {
      wih_b[l][d] = (unsigned short*)alloc((size_t)G4H * din[l] * 2);
      whh_b[l][d] = (unsigned short*)alloc((size_t)G4H * Hq * 2);
    }
  unsigned short* fcw_b = (unsigned short*)alloc((size_t)Kq * (2 * Hq) * 2);

  // allow >64KB dynamic LDS on the scan kernel (320 KB/WGP on CDNA5)
  hipFuncSetAttribute((const void*)lstm_scan_kernel,
                      hipFuncAttributeMaxDynamicSharedMemorySize, SCAN_LDS_BYTES);

  init_ints_kernel<<<1, 64, 0, stream>>>(barrier, 64);
  init_ints_kernel<<<2, 256, 0, stream>>>((int*)zrow, 512);

  // weight conversions fp32 -> bf16
  auto conv = [&](const float* s, unsigned short* d, int n) {
    int blocks = (n + 256 * 16 - 1) / (256 * 16);
    if (blocks > 4096) blocks = 4096;
    f2bf_kernel<<<blocks, 256, 0, stream>>>(s, d, n);
  };
  for (int l = 0; l < 2; ++l)
    for (int d = 0; d < 2; ++d) {
      conv(Wih_f[l][d], wih_b[l][d], G4H * din[l]);
      conv(Whh_f[l][d], whh_b[l][d], G4H * Hq);
    }
  conv(fcW, fcw_b, Kq * 2 * Hq);

  // embedding gather + mask
  embed_kernel<<<MROWS, Eq, 0, stream>>>(ids, emb, xin, maskf);

  // two BiLSTM layers
  for (int l = 0; l < 2; ++l) {
    for (int d = 0; d < 2; ++d) {
      // xg = xin @ W_ih^T + b     [8192, 2048]
      gemm_bf16_kernel<<<dim3(MROWS / 32, G4H / 128), 256, 0, stream>>>(
          xin, 1024, wih_b[l][d], din[l], bias_f[l][d],
          xg, G4H, MROWS, G4H, din[l], zrow);
      // sequential scan (d==1 -> reverse, writes y cols [512,1024))
      lstm_scan_kernel<<<SCAN_BLOCKS, 256, SCAN_LDS_BYTES, stream>>>(
          xg, whh_b[l][d], h0, h1, maskf, ybuf, d ? Hq : 0, d,
          barrier, barrier + 16);
    }
    // layer output (fp32) -> next layer input (bf16)
    f2bf_kernel<<<4096, 256, 0, stream>>>(ybuf, xin, MROWS * 1024);
  }

  // emissions: em = y @ fc_W^T + fc_b    [8192, 24]
  gemm_bf16_kernel<<<dim3(MROWS / 32, 1), 256, 0, stream>>>(
      xin, 1024, fcw_b, 2 * Hq, fcb, embuf, Kq, MROWS, Kq, 2 * Hq, zrow);

  // CRF score + logZ + mean NLL
  crf_kernel<<<1, 1024, 0, stream>>>(embuf, maskf, tags, lengths,
                                     crf_s, crf_e, crf_t, (float*)d_out);
}
